// TimeLSTM_3401614098996
// MI455X (gfx1250) — matrix operand order repaired and checked by reading, not compiled, not run on hardware
//
#include <hip/hip_runtime.h>

// ---------------------------------------------------------------------------
// TimeLSTM fused persistent kernel for MI455X (gfx1250, wave32, WMMA).
//
// Roofline: ~100 MB mandatory HBM traffic (4.3 us floor), ~24 GFLOP total,
// but the scan serializes over SEQ=512 and batch tiling caps parallelism at
// 256/16 = 16 WMMA M-tiles. So: 16 workgroups (one per WGP), all weights
// resident in LDS as bf16 (CDNA5: 320 KB/WGP), both per-step GEMMs done with
// v_wmma_f32_16x16x32_bf16, gate nonlinearities with v_exp_f32.
// ---------------------------------------------------------------------------

typedef __bf16 v16bf __attribute__((ext_vector_type(16)));
typedef float  v8f   __attribute__((ext_vector_type(8)));

#define BATCH 256
#define SEQ   512
#define IND   64
#define HS    128

#define NBLK  16      // batch tiles of 16 rows
#define NTHR  256     // 8 wave32 per workgroup

// ---- LDS layout (bytes), dynamic shared memory ----
// Wh packed bf16: 24 N-tiles x 4 K-chunks x 32 lanes x 16 elems
#define WH_OFF   0                       // 24*4*32*16*2  = 98304
#define WX_OFF   (WH_OFF + 98304)        // 40*2*32*16*2  = 81920
#define G_OFF    (WX_OFF + 81920)        // 16*640*4      = 40960
#define C_OFF    (G_OFF + 40960)         // 16*128*4      = 8192
#define HP_OFF   (C_OFF + 8192)          // 4*32*16*2     = 4096  (h, A-layout)
#define AP_OFF   (HP_OFF + 4096)         // 2*32*16*2     = 2048  (x_t, A-layout)
#define TD_OFF   (AP_OFF + 2048)         // 16*4          = 64
#define BIAS_OFF (TD_OFF + 64)           // 640*4         = 2560
#define WT1_OFF  (BIAS_OFF + 2560)       // 128*4
#define WTA_OFF  (WT1_OFF + 512)
#define WTB_OFF  (WTA_OFF + 512)
#define SMEM_BYTES (WTB_OFF + 512)       // 239680 B  (< 320 KB per WGP)

static __device__ __forceinline__ __bf16 f2bf(float f) {
    unsigned u = __builtin_bit_cast(unsigned, f);
    unsigned r = u + 0x7FFFu + ((u >> 16) & 1u);   // round-to-nearest-even
    unsigned short h = (unsigned short)(r >> 16);
    return __builtin_bit_cast(__bf16, h);
}
static __device__ __forceinline__ float sigm(float x) {
    return 1.0f / (1.0f + __expf(-x));
}
static __device__ __forceinline__ float tanh_pf(float x) {
    float a = __expf(2.0f * fabsf(x));             // overflow-safe tanh
    float t = 1.0f - 2.0f / (a + 1.0f);
    return copysignf(t, x);
}

// A-matrix (16-bit, 16xK) per-lane mapping (ISA 7.12.2): group-of-8 interleave
static __device__ __forceinline__ int a_slot(int M, int K) {
    int chunk = K >> 5, r = K & 31, g = r >> 3, t = r & 7;
    int lane = M + ((g & 1) << 4);
    int e    = t + ((g >> 1) << 3);
    return (chunk * 32 + lane) * 16 + e;
}
// B-matrix (16-bit, Kx16) per-lane mapping: group-of-8 interleave (matches A)
static __device__ __forceinline__ int b_slot(int tile, int tile_stride_chunks,
                                             int K, int N) {
    int chunk = K >> 5, k32 = K & 31;
    int lane  = (N & 15) + (((k32 >> 3) & 1) << 4);
    int e     = (k32 & 7) + ((k32 >> 4) << 3);
    return ((tile * tile_stride_chunks + chunk) * 32 + lane) * 16 + e;
}

__global__ void __launch_bounds__(NTHR)
time_lstm_fused(const float* __restrict__ x, const float* __restrict__ td,
                const float* __restrict__ wx, const float* __restrict__ wh,
                const float* __restrict__ wt1, const float* __restrict__ wt,
                const float* __restrict__ bias, float* __restrict__ out) {
    extern __shared__ char smem[];
    __bf16* whp = (__bf16*)(smem + WH_OFF);
    __bf16* wxp = (__bf16*)(smem + WX_OFF);
    float*  G   = (float*)(smem + G_OFF);
    float*  cls = (float*)(smem + C_OFF);
    __bf16* hp  = (__bf16*)(smem + HP_OFF);
    __bf16* ap  = (__bf16*)(smem + AP_OFF);
    float*  tdl = (float*)(smem + TD_OFF);
    float*  bls = (float*)(smem + BIAS_OFF);
    float*  w1l = (float*)(smem + WT1_OFF);
    float*  wal = (float*)(smem + WTA_OFF);
    float*  wbl = (float*)(smem + WTB_OFF);

    const int tid = threadIdx.x;
    const int b0  = blockIdx.x * 16;

    // ---- one-time init: pack weights into LDS (bf16, WMMA layouts) ----
    for (int idx = tid; idx < HS * 384; idx += NTHR) {       // Wh: (128,384)
        int k = idx / 384, n = idx - k * 384;
        whp[b_slot(n >> 4, 4, k, n)] = f2bf(wh[idx]);
    }
    for (int idx = tid; idx < IND * 640; idx += NTHR) {      // Wx: (64,640)
        int k = idx / 640, n = idx - k * 640;
        wxp[b_slot(n >> 4, 2, k, n)] = f2bf(wx[idx]);
    }
    for (int idx = tid; idx < 640; idx += NTHR) bls[idx] = bias[idx];
    if (tid < HS) {
        w1l[tid] = fminf(wt1[tid], 0.0f);   // wt1 = min(weights_t1, 0)
        wal[tid] = wt[tid];                 // weights_t[:, :HS]
        wbl[tid] = wt[HS + tid];            // weights_t[:, HS:]
    }
    for (int idx = tid; idx < 16 * HS; idx += NTHR) cls[idx] = 0.0f;   // c0 = 0
    for (int idx = tid; idx < 4 * 32 * 16; idx += NTHR)                 // h0 = 0
        hp[idx] = __builtin_bit_cast(__bf16, (unsigned short)0);
    __syncthreads();

    const int wave  = tid >> 5;
    const int lane  = tid & 31;
    const int nloc  = lane & 15;
    const int mbase = (lane >> 4) << 3;     // C/D: M = r + mbase

    float* out_h  = out;                                 // (B,S,HS)
    float* out_hT = out + (size_t)BATCH * SEQ * HS;      // (B,HS)
    float* out_cT = out_hT + BATCH * HS;                 // (B,HS)

    for (int s = 0; s < SEQ; ++s) {
        // ---- stage x_t (16x64) -> bf16 A-layout; TimeDiff column ----
        {
            int b  = tid >> 4;
            int kb = (tid & 15) << 2;
            const float4 xv = *reinterpret_cast<const float4*>(
                x + ((size_t)(b0 + b) * SEQ + s) * IND + kb);
            float xs[4] = {xv.x, xv.y, xv.z, xv.w};
#pragma unroll
            for (int q = 0; q < 4; ++q)
                ap[a_slot(b, kb + q)] = f2bf(xs[q]);
            if (tid < 16) tdl[tid] = td[(size_t)(b0 + tid) * SEQ + s];
        }
        __syncthreads();

        // ---- GEMM1: x_t @ Wx -> G[16][640]  (5 N-tiles/wave, K=64) ----
        const v16bf* av  = reinterpret_cast<const v16bf*>(ap);
        const v16bf* wxv = reinterpret_cast<const v16bf*>(wxp);
#pragma unroll
        for (int q = 0; q < 5; ++q) {
            int tile = wave * 5 + q;
            v8f acc = {};
#pragma unroll
            for (int ch = 0; ch < 2; ++ch) {
                v16bf a = av[ch * 32 + lane];
                v16bf b = wxv[((tile << 1) + ch) * 32 + lane];
                acc = __builtin_amdgcn_wmma_f32_16x16x32_bf16(
                    false, a, false, b, (short)0, acc, false, false);
            }
            int colb = tile * 16 + nloc;
#pragma unroll
            for (int r = 0; r < 8; ++r)
                G[(r + mbase) * 640 + colb] = acc[r];
        }
        __syncthreads();

        // ---- GEMM2: h @ Wh -> += into gate i/c/o columns of G (K=128) ----
        const v16bf* hv  = reinterpret_cast<const v16bf*>(hp);
        const v16bf* whv = reinterpret_cast<const v16bf*>(whp);
#pragma unroll
        for (int q = 0; q < 3; ++q) {
            int tile = wave * 3 + q;
            v8f acc = {};
#pragma unroll
            for (int ch = 0; ch < 4; ++ch) {
                v16bf a = hv[ch * 32 + lane];
                v16bf b = whv[((tile << 2) + ch) * 32 + lane];
                acc = __builtin_amdgcn_wmma_f32_16x16x32_bf16(
                    false, a, false, b, (short)0, acc, false, false);
            }
            int nW = tile * 16 + nloc;             // Wh col: 0..383
            // Wh_i -> G[0:128), Wh_c -> G[384:512), Wh_o -> G[512:640)
            int gcol = nW + ((nW >= 128) ? 256 : 0);
#pragma unroll
            for (int r = 0; r < 8; ++r)
                G[(r + mbase) * 640 + gcol] += acc[r];
        }
        __syncthreads();

        // ---- elementwise gates, state update, h repack (8 elems/thread) ----
#pragma unroll
        for (int p = 0; p < 8; ++p) {
            int idx = p * NTHR + tid;      // 0..2047
            int M = idx >> 7, j = idx & 127;
            float tdv = tdl[M];
            float pi  = G[M * 640 + j]       + bls[j];
            float pt1 = G[M * 640 + 128 + j] + bls[128 + j] + tanh_pf(tdv * w1l[j]);
            float pt2 = G[M * 640 + 256 + j] + bls[256 + j] + tanh_pf(tdv * wal[j]);
            float pc  = G[M * 640 + 384 + j] + bls[384 + j];
            float po  = G[M * 640 + 512 + j] + bls[512 + j] + tdv * wbl[j];
            float i_t = sigm(pi);
            float t1  = sigm(pt1);
            float t2  = sigm(pt2);
            float cw  = tanh_pf(pc);
            float o_t = sigm(po);
            float c   = cls[idx];
            float it1 = i_t * t1;
            float ctl = sigm((1.0f - it1) * c + it1 * cw);
            float cnw = sigm((1.0f - i_t) * c + i_t * t2 * cw);
            float hnw = o_t + tanh_pf(ctl);
            cls[idx] = cnw;
            __builtin_nontemporal_store(
                hnw, &out_h[((size_t)(b0 + M) * SEQ + s) * HS + j]);
            hp[a_slot(M, j)] = f2bf(hnw);          // h_{t} as next-step A matrix
            if (s == SEQ - 1) {
                out_hT[(b0 + M) * HS + j] = hnw;
                out_cT[(b0 + M) * HS + j] = cnw;
            }
        }
        __syncthreads();   // tdl/G/hp reuse next iteration
    }
}

extern "C" void kernel_launch(void* const* d_in, const int* in_sizes, int n_in,
                              void* d_out, int out_size, void* d_ws, size_t ws_size,
                              hipStream_t stream) {
    (void)in_sizes; (void)n_in; (void)out_size; (void)d_ws; (void)ws_size;
    const float* x    = (const float*)d_in[0];
    const float* tdif = (const float*)d_in[1];
    const float* wx   = (const float*)d_in[2];
    const float* wh   = (const float*)d_in[3];
    const float* wt1  = (const float*)d_in[4];
    const float* wt   = (const float*)d_in[5];
    const float* bias = (const float*)d_in[6];
    float* out = (float*)d_out;

    // Allow >64KB dynamic LDS (CDNA5 WGP has 320 KB). Idempotent, capture-safe.
    hipFuncSetAttribute(reinterpret_cast<const void*>(time_lstm_fused),
                        hipFuncAttributeMaxDynamicSharedMemorySize, SMEM_BYTES);
    time_lstm_fused<<<NBLK, NTHR, SMEM_BYTES, stream>>>(x, tdif, wx, wh, wt1,
                                                        wt, bias, out);
}
